// Encoding_51771535786498
// MI455X (gfx1250) — matrix operand order repaired and checked by reading, not compiled, not run on hardware
//
#include <hip/hip_runtime.h>
#include <hip/hip_bf16.h>

// ---------------------------------------------------------------------------
// EncNet "Encoding" layer for MI455X (gfx1250, wave32, WMMA, TDM).
// B=64, C=512, H=W=28 -> N=784, K=32.
//
//  k_prep      : codewords -> bf16 (ws), c_sq[k] = sum_c cw[k,c]^2
//  k_assign    : TDM-stage cw_bf into LDS; per-wave 32k x 16n tile of
//                xc = cw @ x (WMMA bf16), fused x_sq, softmax over K,
//                store A as (b,k,n) bf16 in ws
//  k_aggregate : TDM-stage A[b] (49 KB) into LDS; per-wave 32k x 16c tile of
//                wx = A @ x^T (WMMA bf16); wsum via ones-matrix WMMA;
//                output = wx - wsum*cw (f32)
// ---------------------------------------------------------------------------

typedef __attribute__((ext_vector_type(16))) __bf16 v16bf;
typedef __attribute__((ext_vector_type(2)))  __bf16 v2bf;
typedef __attribute__((ext_vector_type(8)))  float  v8f;
typedef __attribute__((ext_vector_type(4)))  unsigned int uint4v;
typedef __attribute__((ext_vector_type(8)))  int    int8v;
typedef __attribute__((ext_vector_type(4)))  int    int4v;

union BF16x16 {
    v16bf v;
    unsigned int w[8];
};

__device__ __forceinline__ unsigned int pack2bf(float a, float b) {
    v2bf t;
    t.x = (__bf16)a;   // RNE -> v_cvt_pk_bf16_f32
    t.y = (__bf16)b;
    return __builtin_bit_cast(unsigned int, t);
}
__device__ __forceinline__ unsigned short f2bfu(float f) {
    return __builtin_bit_cast(unsigned short, (__bf16)f);
}

#define ENC_B 64
#define ENC_C 512
#define ENC_N 784
#define ENC_K 32

// ---------------------------------------------------------------------------
// TDM: 1-D DMA of n2 2-byte elements from global -> LDS (issue from one wave).
// Descriptor bitfields per CDNA5 ISA ch.8 (D# group0/group1), groups 2/3/4 zero.
// ---------------------------------------------------------------------------
__device__ __forceinline__ void tdm_load_2b(const void* gptr, unsigned lds_off,
                                            unsigned n2) {
    unsigned long long ga = (unsigned long long)(uintptr_t)gptr;
    uint4v g0;
    g0[0] = 1u;                                            // count=1, user mode
    g0[1] = lds_off;                                       // lds_addr (bytes)
    g0[2] = (unsigned)ga;                                  // global_addr[31:0]
    g0[3] = (unsigned)((ga >> 32) & 0x1FFFFFFu) | (2u << 30); // addr[56:32]|type=2
    int8v g1;
    g1[0] = (int)(1u << 16);                               // data_size=1 (2B), wg_mask=0
    g1[1] = (int)((n2 & 0xFFFFu) << 16);                   // tensor_dim0[15:0]
    g1[2] = (int)(((n2 >> 16) & 0xFFFFu) | (1u << 16));    // tensor_dim0[31:16], tensor_dim1=1
    g1[3] = (int)((n2 & 0xFFFFu) << 16);                   // tile_dim0 = n2
    g1[4] = 1;                                             // tile_dim1=1, tile_dim2=0
    g1[5] = (int)n2;                                       // tensor_dim0_stride[31:0]
    g1[6] = 0;                                             // stride hi, dim1_stride lo
    g1[7] = 0;
    int4v gz4 = {0, 0, 0, 0};
    int8v gz8 = {0, 0, 0, 0, 0, 0, 0, 0};
    __builtin_amdgcn_tensor_load_to_lds(g0, g1, gz4, gz4, gz8, 0);
}

// ---------------------------------------------------------------------------
// Kernel 0: codewords f32 -> bf16, and c_sq.
// ---------------------------------------------------------------------------
__global__ __launch_bounds__(256) void k_prep(const float* __restrict__ cw,
                                              unsigned short* __restrict__ cw_bf,
                                              float* __restrict__ c_sq) {
    int t = threadIdx.x;
    for (int i = t; i < ENC_K * ENC_C; i += 256) cw_bf[i] = f2bfu(cw[i]);
    if (t < ENC_K) {
        float s = 0.f;
        const float* row = cw + t * ENC_C;
        for (int c = 0; c < ENC_C; ++c) { float v = row[c]; s += v * v; }
        c_sq[t] = s;
    }
}

// ---------------------------------------------------------------------------
// Kernel 1: assignment (xc GEMM + softmax). grid = B*7, block = 224 (7 waves).
// ---------------------------------------------------------------------------
__global__ __launch_bounds__(224) void k_assign(const float* __restrict__ x,
                                                const float* __restrict__ scale,
                                                const unsigned short* __restrict__ cw_bf,
                                                const float* __restrict__ c_sq,
                                                unsigned short* __restrict__ Amat) {
    __shared__ __align__(16) unsigned short shCw[ENC_K * ENC_C];  // 32 KB
    __shared__ float sScale[ENC_K];
    __shared__ float sCsq[ENC_K];

    if (threadIdx.x < 32) {
        // one wave stages codewords via TDM, also loads scale/c_sq
        tdm_load_2b(cw_bf, (unsigned)(uintptr_t)(void*)shCw, ENC_K * ENC_C);
        sScale[threadIdx.x] = scale[threadIdx.x];
        sCsq[threadIdx.x]   = c_sq[threadIdx.x];
        __builtin_amdgcn_s_wait_tensorcnt(0);
    }
    __syncthreads();

    const int b     = blockIdx.x / 7;
    const int tgrp  = blockIdx.x % 7;
    const int wave  = threadIdx.x >> 5;
    const int lane  = threadIdx.x & 31;
    const int ntile = tgrp * 7 + wave;          // 0..48
    const int n0    = ntile * 16;
    const int col   = lane & 15;                // n column within tile
    const int half  = lane >> 4;
    const int ncol  = n0 + col;
    const int row   = col;                      // k row for A-operand lanes

    const float* __restrict__ xb = x + (size_t)b * ENC_C * ENC_N;

    v8f acc0 = {0.f, 0.f, 0.f, 0.f, 0.f, 0.f, 0.f, 0.f};
    v8f acc1 = {0.f, 0.f, 0.f, 0.f, 0.f, 0.f, 0.f, 0.f};
    float xs0 = 0.f, xs1 = 0.f, xs2 = 0.f, xs3 = 0.f;  // independent chains

#pragma unroll 2
    for (int c0 = 0; c0 < ENC_C; c0 += 32) {
        // speculative prefetch of the next c-tile (silently dropped past end)
        __builtin_prefetch(xb + (c0 + 32 + 16 * half) * ENC_N + ncol, 0, 1);
        BF16x16 bmat, a0, a1;
#pragma unroll
        for (int v = 0; v < 8; ++v) {
            // B operand (32c x 16n): lane col = n, vgpr v holds c = c0+2v+16*half+{0,1}
            int c = c0 + 2 * v + 16 * half;
            float e0 = xb[c * ENC_N + ncol];
            float e1 = xb[(c + 1) * ENC_N + ncol];
            if (v & 1) { xs2 += e0 * e0; xs3 += e1 * e1; }
            else       { xs0 += e0 * e0; xs1 += e1 * e1; }
            bmat.w[v] = pack2bf(e0, e1);
            // A operand (16k x 32c): vgpr v -> cdim 2v+16*half
            int cc = c0 + 2 * v + 16 * half;
            a0.w[v] = *(const unsigned int*)&shCw[row * ENC_C + cc];
            a1.w[v] = *(const unsigned int*)&shCw[(row + 16) * ENC_C + cc];
        }
        acc0 = __builtin_amdgcn_wmma_f32_16x16x32_bf16(false, a0.v, false, bmat.v,
                                                       (short)0, acc0, false, false);
        acc1 = __builtin_amdgcn_wmma_f32_16x16x32_bf16(false, a1.v, false, bmat.v,
                                                       (short)0, acc1, false, false);
    }

    // x_sq[n]: lane pair {n, n+16} covers complementary c halves
    float xsq = (xs0 + xs1) + (xs2 + xs3);
    xsq += __shfl_xor(xsq, 16, 32);

    // D layout: lane col = lane&15; rows m = 8*half + r  (acc0: k=m, acc1: k=16+m)
    float d0[8], d1[8];
    float mx = -3.4e38f;
#pragma unroll
    for (int r = 0; r < 8; ++r) {
        int k0 = 8 * half + r;
        int k1 = 16 + k0;
        d0[r] = sScale[k0] * (xsq + sCsq[k0] - 2.0f * acc0[r]);
        d1[r] = sScale[k1] * (xsq + sCsq[k1] - 2.0f * acc1[r]);
        mx = fmaxf(mx, fmaxf(d0[r], d1[r]));
    }
    mx = fmaxf(mx, __shfl_xor(mx, 16, 32));

    float sum = 0.f;
#pragma unroll
    for (int r = 0; r < 8; ++r) {
        d0[r] = __expf(d0[r] - mx);
        d1[r] = __expf(d1[r] - mx);
        sum += d0[r] + d1[r];
    }
    sum += __shfl_xor(sum, 16, 32);
    float inv = __frcp_rn(sum);

    unsigned short* __restrict__ Ab = Amat + (size_t)b * ENC_K * ENC_N;
#pragma unroll
    for (int r = 0; r < 8; ++r) {
        int k0 = 8 * half + r;
        Ab[k0 * ENC_N + ncol]        = f2bfu(d0[r] * inv);
        Ab[(16 + k0) * ENC_N + ncol] = f2bfu(d1[r] * inv);
    }
}

// ---------------------------------------------------------------------------
// Kernel 2: aggregation (wx GEMM + wsum + output). grid = B*8, block = 128.
// A[b] staged in LDS via TDM; wsum computed with ones-matrix WMMA.
// ---------------------------------------------------------------------------
__global__ __launch_bounds__(128) void k_aggregate(const float* __restrict__ x,
                                                   const float* __restrict__ cw,
                                                   const unsigned short* __restrict__ Amat,
                                                   float* __restrict__ out) {
    __shared__ __align__(16) unsigned short shA[ENC_K * ENC_N];  // 49 KB

    const int b    = blockIdx.x >> 3;
    const int ct   = blockIdx.x & 7;
    const int wave = threadIdx.x >> 5;
    const int lane = threadIdx.x & 31;
    const int c0   = ct * 64 + wave * 16;
    const int col  = lane & 15;                 // c column within tile
    const int half = lane >> 4;
    const int c    = c0 + col;
    const int row  = lane & 15;                 // k row for A-operand lanes

    if (threadIdx.x < 32) {
        tdm_load_2b(Amat + (size_t)b * ENC_K * ENC_N,
                    (unsigned)(uintptr_t)(void*)shA, ENC_K * ENC_N);
        __builtin_amdgcn_s_wait_tensorcnt(0);
    }
    __syncthreads();

    const float* __restrict__ xb = x + (size_t)b * ENC_C * ENC_N;

    v8f acc0 = {0.f, 0.f, 0.f, 0.f, 0.f, 0.f, 0.f, 0.f};
    v8f acc1 = {0.f, 0.f, 0.f, 0.f, 0.f, 0.f, 0.f, 0.f};
    v8f sum0 = {0.f, 0.f, 0.f, 0.f, 0.f, 0.f, 0.f, 0.f};  // ones-GEMM: wsum k=0..15
    v8f sum1 = {0.f, 0.f, 0.f, 0.f, 0.f, 0.f, 0.f, 0.f};  // ones-GEMM: wsum k=16..31

    BF16x16 bones;
#pragma unroll
    for (int v = 0; v < 8; ++v) bones.w[v] = 0x3F803F80u;  // bf16 1.0 pairs

    // 24 full steps of 32 over n
#pragma unroll 2
    for (int n0 = 0; n0 < 768; n0 += 32) {
        BF16x16 a0, a1, bmat;
#pragma unroll
        for (int v = 0; v < 8; ++v) {
            // A operand (16k x 32n) from LDS
            int nn = n0 + 2 * v + 16 * half;
            a0.w[v] = *(const unsigned int*)&shA[row * ENC_N + nn];
            a1.w[v] = *(const unsigned int*)&shA[(row + 16) * ENC_N + nn];
            // B operand (32n x 16c): lane col = c, vgpr v holds n = n0+2v+16*half+{0,1}
            int nb = n0 + 2 * v + 16 * half;
            float2 p = *(const float2*)(xb + c * ENC_N + nb);
            bmat.w[v] = pack2bf(p.x, p.y);
        }
        acc0 = __builtin_amdgcn_wmma_f32_16x16x32_bf16(false, a0.v, false, bmat.v,
                                                       (short)0, acc0, false, false);
        acc1 = __builtin_amdgcn_wmma_f32_16x16x32_bf16(false, a1.v, false, bmat.v,
                                                       (short)0, acc1, false, false);
        sum0 = __builtin_amdgcn_wmma_f32_16x16x32_bf16(false, a0.v, false, bones.v,
                                                       (short)0, sum0, false, false);
        sum1 = __builtin_amdgcn_wmma_f32_16x16x32_bf16(false, a1.v, false, bones.v,
                                                       (short)0, sum1, false, false);
    }

    // tail: n0 = 768, valid n in [768, 784)
    {
        const int n0 = 768;
        BF16x16 a0, a1, bmat;
#pragma unroll
        for (int v = 0; v < 8; ++v) {
            if (half == 0) {  // nn = 768 + 2v <= 782 : valid
                int nn = n0 + 2 * v + 16 * half;
                a0.w[v] = *(const unsigned int*)&shA[row * ENC_N + nn];
                a1.w[v] = *(const unsigned int*)&shA[(row + 16) * ENC_N + nn];
            } else {      // nn >= 784 : pad with zeros
                a0.w[v] = 0u;
                a1.w[v] = 0u;
            }
            int nb = n0 + 2 * v + 16 * half;
            if (half == 0) {  // n in 768..783 : valid
                float2 p = *(const float2*)(xb + c * ENC_N + nb);
                bmat.w[v] = pack2bf(p.x, p.y);
            } else {          // n >= 784 : pad
                bmat.w[v] = 0u;
            }
        }
        acc0 = __builtin_amdgcn_wmma_f32_16x16x32_bf16(false, a0.v, false, bmat.v,
                                                       (short)0, acc0, false, false);
        acc1 = __builtin_amdgcn_wmma_f32_16x16x32_bf16(false, a1.v, false, bmat.v,
                                                       (short)0, acc1, false, false);
        sum0 = __builtin_amdgcn_wmma_f32_16x16x32_bf16(false, a0.v, false, bones.v,
                                                       (short)0, sum0, false, false);
        sum1 = __builtin_amdgcn_wmma_f32_16x16x32_bf16(false, a1.v, false, bones.v,
                                                       (short)0, sum1, false, false);
    }

    // Output: D layout lane col = c, rows m = 8*half + r; k = m (acc0), 16+m (acc1).
    // sum0[r] = wsum[k=m], sum1[r] = wsum[k=16+m] -- already in the right lane.
    float* __restrict__ ob = out + (size_t)b * ENC_K * ENC_C;
#pragma unroll
    for (int r = 0; r < 8; ++r) {
        int m = 8 * half + r;
        ob[m * ENC_C + c]        = acc0[r] - sum0[r] * cw[m * ENC_C + c];
        ob[(16 + m) * ENC_C + c] = acc1[r] - sum1[r] * cw[(16 + m) * ENC_C + c];
    }
}

// ---------------------------------------------------------------------------
// Launcher
// ---------------------------------------------------------------------------
extern "C" void kernel_launch(void* const* d_in, const int* in_sizes, int n_in,
                              void* d_out, int out_size, void* d_ws, size_t ws_size,
                              hipStream_t stream) {
    const float* x     = (const float*)d_in[0];   // (64, 512, 28, 28)
    const float* cw    = (const float*)d_in[1];   // (32, 512)
    const float* scale = (const float*)d_in[2];   // (32,)
    float* out = (float*)d_out;                   // (64, 32, 512)

    char* ws = (char*)d_ws;
    unsigned short* cw_bf = (unsigned short*)ws;                 // 32 KB
    float*          c_sq  = (float*)(ws + 32768);                // 128 B
    unsigned short* Amat  = (unsigned short*)(ws + 40960);       // 64*32*784*2 B

    k_prep<<<1, 256, 0, stream>>>(cw, cw_bf, c_sq);
    k_assign<<<ENC_B * 7, 224, 0, stream>>>(x, scale, cw_bf, c_sq, Amat);
    k_aggregate<<<ENC_B * 8, 128, 0, stream>>>(x, cw, Amat, out);
}
